// Posterior3D_spec_87273735455398
// MI455X (gfx1250) — matrix operand order repaired and checked by reading, not compiled, not run on hardware
//
#include <hip/hip_runtime.h>
#include <hip/hip_bf16.h>
#include <math.h>

#define NGRID    8192
#define NLAYER   24
#define NPERIODS 50
#define NOBS     4096
#define NC       200
#define NPAD     64          // periods padded to 4 WMMA N-tiles
#define RAYLEIGH 0.92f
#define VREF     3.0f
#define KSEG     16          // K split across gridDim.y
#define KCHUNK   (NGRID / KSEG)   // 512

typedef __attribute__((ext_vector_type(16))) __bf16 v16bf;
typedef __attribute__((ext_vector_type(8)))  float  v8f;
typedef __attribute__((ext_vector_type(4)))  float  v4f;

// ---------------------------------------------------------------------------
// Workspace layout:
//   [0       , 1 MiB) : SinvT  bf16  [NPAD][NGRID]   (B operand, transposed)
//   [1 MiB   , 2 MiB) : s_pred f32   [NOBS][NPAD]
// ---------------------------------------------------------------------------

__global__ void k_init(float* __restrict__ s_pred, float* __restrict__ out) {
    int i = blockIdx.x * blockDim.x + threadIdx.x;
    if (i < NOBS * NPAD) s_pred[i] = 0.0f;
    if (i == 0) out[0] = 0.0f;
}

// ---------------------------------------------------------------------------
// Kernel 1: surrogate dispersion -> SinvT[p][g] = bf16( 1 / c_true[g][p] )
// One block of 256 threads covers 256 consecutive g for a single p
// (NGRID/256 = 32 blocks per p). Vs slab staged coalesced into LDS with
// stride-25 padding (gcd(25,64)=1 -> conflict-free dot-product reads).
// ---------------------------------------------------------------------------
__global__ void __launch_bounds__(256)
k_dispersion(const float* __restrict__ Vs,
             const float* __restrict__ thick,
             const float* __restrict__ periods,
             __bf16* __restrict__ SinvT) {
    __shared__ float sw[NLAYER];
    __shared__ float swsum;
    __shared__ float svs[256 * (NLAYER + 1)];        // padded stride 25

    const int p = (blockIdx.x * 256) >> 13;          // constant within block
    const int gbase = (blockIdx.x * 256) & (NGRID - 1);
    const float period = (p < NPERIODS) ? periods[p] : 1.0f;

    if (threadIdx.x < NLAYER) {
        float zc = 0.0f;
        for (int l = 0; l <= (int)threadIdx.x; ++l) zc += thick[l];
        const float z  = zc - 0.5f * thick[threadIdx.x];   // layer-center depth
        const float ds = VREF * period * (1.0f / 3.0f);    // sensing depth
        sw[threadIdx.x] = __expf(-z / ds) * thick[threadIdx.x];
    }
    if (p < NPERIODS) {                               // coalesced LDS stage
        for (int idx = threadIdx.x; idx < 256 * NLAYER; idx += 256) {
            const int gl = idx / NLAYER;
            const int l  = idx - gl * NLAYER;
            svs[gl * (NLAYER + 1) + l] = Vs[(size_t)gbase * NLAYER + idx];
        }
    }
    __syncthreads();
    if (threadIdx.x == 0) {
        float s = 0.0f;
        for (int l = 0; l < NLAYER; ++l) s += sw[l];
        swsum = s;
    }
    __syncthreads();

    const int g = gbase + threadIdx.x;

    __bf16 outv = (__bf16)0.0f;                       // N-padding -> zero cols
    if (p < NPERIODS) {
        const float* vr = svs + threadIdx.x * (NLAYER + 1);
        float dot = 0.0f;
        #pragma unroll
        for (int l = 0; l < NLAYER; ++l) dot = fmaf(sw[l], vr[l], dot);
        const float c_true = RAYLEIGH * dot / swsum;
        outv = (__bf16)(1.0f / c_true);
    }
    SinvT[(size_t)p * NGRID + g] = outv;              // coalesced over g
}

// ---------------------------------------------------------------------------
// Kernel 2: s_pred = A @ Sinv  via v_wmma_f32_16x16x32_bf16.
//   blockIdx.x = M-tile (0..255), blockIdx.y = K segment (0..15), 1 wave/WG.
//   A fragment loaded once per K-step and reused across 4 N-tiles; partials
//   combined with global_atomic_add_f32 into zero-initialized s_pred.
// ---------------------------------------------------------------------------
__global__ void __launch_bounds__(32)
k_gemm_wmma(const float* __restrict__ A,
            const __bf16* __restrict__ SinvT,
            float* __restrict__ s_pred) {
    const int lane = threadIdx.x;          // 0..31 (wave32)
    const int m    = lane & 15;            // row within tile (both halves)
    const int h    = lane >> 4;            // K-half selector (ISA A/B layouts)
    const int mtile = blockIdx.x;
    const int kbeg  = blockIdx.y * KCHUNK;
    const int kend  = kbeg + KCHUNK;

    const size_t arow = (size_t)(mtile * 16 + m) * NGRID;

    v8f acc0 = {}, acc1 = {}, acc2 = {}, acc3 = {};

    for (int kb = kbeg; kb < kend; kb += 32) {
        // ---- A fragment: 16-bit A-matrix 16x32 layout ----
        // lane (m,h): element s -> K = h*16 + s (matches B fragment)
        const float* ap = A + arow + kb + h * 16;
        __builtin_prefetch(ap + 128, 0, 1);            // global_prefetch_b8
        v4f r0 = *(const v4f*)(ap);
        v4f r1 = *(const v4f*)(ap + 4);
        v4f r2 = *(const v4f*)(ap + 8);
        v4f r3 = *(const v4f*)(ap + 12);
        v16bf a;
        #pragma unroll
        for (int e = 0; e < 4; ++e) {
            a[e]      = (__bf16)r0[e];
            a[4 + e]  = (__bf16)r1[e];
            a[8 + e]  = (__bf16)r2[e];
            a[12 + e] = (__bf16)r3[e];
        }

        // ---- B fragments: 16-bit B-matrix 32x16 layout ----
        // lane half h holds K = h*16 .. h*16+15 of column n = lane&15,
        // i.e. 16 consecutive bf16 of a SinvT row -> one 32B load per N-tile.
        const __bf16* bbase = SinvT + (size_t)m * NGRID + kb + h * 16;
        v16bf b0 = *(const v16bf*)(bbase);
        v16bf b1 = *(const v16bf*)(bbase + 16 * (size_t)NGRID);
        v16bf b2 = *(const v16bf*)(bbase + 32 * (size_t)NGRID);
        v16bf b3 = *(const v16bf*)(bbase + 48 * (size_t)NGRID);

        acc0 = __builtin_amdgcn_wmma_f32_16x16x32_bf16(false, a, false, b0,
                                                       (short)0, acc0, false, false);
        acc1 = __builtin_amdgcn_wmma_f32_16x16x32_bf16(false, a, false, b1,
                                                       (short)0, acc1, false, false);
        acc2 = __builtin_amdgcn_wmma_f32_16x16x32_bf16(false, a, false, b2,
                                                       (short)0, acc2, false, false);
        acc3 = __builtin_amdgcn_wmma_f32_16x16x32_bf16(false, a, false, b3,
                                                       (short)0, acc3, false, false);
    }

    // ---- C/D layout: VGPR r, lanes 0-15 -> M=r, lanes 16-31 -> M=r+8; N=lane&15
    float* dst = s_pred + (size_t)(mtile * 16 + 8 * h) * NPAD + (lane & 15);
    #pragma unroll
    for (int r = 0; r < 8; ++r) {
        atomicAdd(dst + (size_t)r * NPAD +  0, acc0[r]);
        atomicAdd(dst + (size_t)r * NPAD + 16, acc1[r]);
        atomicAdd(dst + (size_t)r * NPAD + 32, acc2[r]);
        atomicAdd(dst + (size_t)r * NPAD + 48, acc3[r]);
    }
}

// ---------------------------------------------------------------------------
// Kernel 3: one WAVE per (obs,period) pair. Lanes scan energy[pair][:]
// coalesced (lane, lane+32, ...) for e_max, wave-reduce via shfl_xor;
// lane 0 does searchsorted + lerp. 8 pairs per 256-thread block.
// ---------------------------------------------------------------------------
__global__ void __launch_bounds__(256)
k_loglike(const float* __restrict__ s_pred,
          const float* __restrict__ energy,
          const float* __restrict__ c_axis,
          float* __restrict__ out) {
    const int lane = threadIdx.x & 31;
    const int wid  = threadIdx.x >> 5;
    const int pair = blockIdx.x * 8 + wid;           // pair = o*NPERIODS + p

    float contrib = 0.0f;
    if (pair < NOBS * NPERIODS) {
        const float* en = energy + (size_t)pair * NC;

        float em = -3.4e38f;
        #pragma unroll
        for (int c = lane; c < NC; c += 32)          // coalesced 128B lines
            em = fmaxf(em, en[c]);
        #pragma unroll
        for (int off = 16; off > 0; off >>= 1)       // wave32 max-reduce
            em = fmaxf(em, __shfl_xor(em, off, 32));

        if (lane == 0) {
            const int o = pair / NPERIODS;
            const int p = pair - o * NPERIODS;
            const float cp = 1.0f / s_pred[(size_t)o * NPAD + p];
            const float* ca = c_axis + (size_t)o * NC;

            // searchsorted (side='left'): first idx with ca[idx] >= cp
            int lo = 0, hi = NC;
            while (lo < hi) {
                int mid = (lo + hi) >> 1;
                if (ca[mid] < cp) lo = mid + 1; else hi = mid;
            }
            int idx = lo;
            idx = (idx < 1) ? 1 : ((idx > NC - 1) ? NC - 1 : idx);

            const float c0 = ca[idx - 1], c1 = ca[idx];
            const float e0 = en[idx - 1], e1 = en[idx];
            const float w  = (cp - c0) / (c1 - c0 + 1e-12f);
            contrib = em - (e0 + w * (e1 - e0));
        }
    }

    __shared__ float red[8];
    if (lane == 0) red[wid] = contrib;
    __syncthreads();
    if (threadIdx.x == 0) {
        float s = 0.0f;
        #pragma unroll
        for (int w = 0; w < 8; ++w) s += red[w];
        atomicAdd(out, -s);                          // SIGMA == 1
    }
}

// ---------------------------------------------------------------------------
extern "C" void kernel_launch(void* const* d_in, const int* in_sizes, int n_in,
                              void* d_out, int out_size, void* d_ws, size_t ws_size,
                              hipStream_t stream) {
    const float* Vs      = (const float*)d_in[0];   // (NGRID*NLAYER,)
    const float* A       = (const float*)d_in[1];   // (NOBS, NGRID)
    const float* energy  = (const float*)d_in[2];   // (NOBS, NPERIODS, NC)
    const float* c_axis  = (const float*)d_in[3];   // (NOBS, NC)
    const float* thick   = (const float*)d_in[4];   // (NLAYER,)
    const float* periods = (const float*)d_in[5];   // (NPERIODS,)

    __bf16* SinvT  = (__bf16*)d_ws;                          // 1 MiB
    float*  s_pred = (float*)((char*)d_ws + (1u << 20));     // 1 MiB
    float*  outp   = (float*)d_out;

    k_init<<<(NOBS * NPAD + 255) / 256, 256, 0, stream>>>(s_pred, outp);

    k_dispersion<<<(NPAD * NGRID) / 256, 256, 0, stream>>>(Vs, thick, periods, SinvT);

    dim3 gg(NOBS / 16, KSEG);                                // 256 x 16 waves
    k_gemm_wmma<<<gg, 32, 0, stream>>>(A, SinvT, s_pred);

    k_loglike<<<(NOBS * NPERIODS) / 8, 256, 0, stream>>>(s_pred, energy, c_axis, outp);
}
